// model_76991583748341
// MI455X (gfx1250) — matrix operand order repaired and checked by reading, not compiled, not run on hardware
//
#include <hip/hip_runtime.h>
#include <math.h>

typedef _Float16 f16;
typedef _Float16 v16h __attribute__((ext_vector_type(16)));
typedef _Float16 v8h  __attribute__((ext_vector_type(8)));
typedef float    v8f  __attribute__((ext_vector_type(8)));

#define N_PART 8192
#define D_DIM  64
#define P_DIM  32
#define H_DIM  512
#define M_STEPS 50
#define KPAD   128   // 1(s) + 64(X) + 32(Yt) = 97, padded to 128
#define HSTR   40    // padded h-staging row stride (f16)
#define DT_F   0.02f
#define SQRT_DT_F 0.14142135623730951f

// ---- branch-free tanh: native v_tanh_f32 if available, else exp2+rcp ----
__device__ __forceinline__ float fast_tanh(float x) {
#if __has_builtin(__builtin_amdgcn_tanhf)
    return __builtin_amdgcn_tanhf(x);
#elif __has_builtin(__builtin_amdgcn_tanh_f32)
    return __builtin_amdgcn_tanh_f32(x);
#else
    float ax = __builtin_fabsf(x);
    float e  = __builtin_amdgcn_exp2f(ax * -2.8853900817779268f);   // exp(-2|x|)
    float t  = (1.0f - e) * __builtin_amdgcn_rcpf(1.0f + e);
    return __builtin_copysignf(t, x);
#endif
}

// A-matrix fragment (16x32 f16): lane l -> row l%16.
// lanes 0-15: VGPR0-3 = K 0..7,  VGPR4-7 = K 8..15
// lanes16-31: VGPR0-3 = K 16..23, VGPR4-7 = K 24..31
__device__ __forceinline__ v16h a_frag(const f16* rowbase, int k0) {
    v8h lo = *(const v8h*)(rowbase + k0);
    v8h hi = *(const v8h*)(rowbase + k0 + 8);
    return __builtin_shufflevector(lo, hi, 0,1,2,3,4,5,6,7,8,9,10,11,12,13,14,15);
}

// Async copy: 8 x b128 per wave moves one 16x64 f32 noise tile (4 KB) to LDS.
__device__ __forceinline__ void noise_prefetch(const float* gbase, unsigned ldsbase, int lane) {
    #pragma unroll
    for (int i = 0; i < 8; ++i) {
        unsigned l = ldsbase + (unsigned)((i * 32 + lane) * 16);
        const float* g = gbase + (i * 32 + lane) * 4;
        asm volatile("global_load_async_to_lds_b128 %0, %1, off"
                     :: "v"(l), "v"((unsigned long long)(uintptr_t)g)
                     : "memory");
    }
}

__global__ __launch_bounds__(128, 1) void sde_wmma_kernel(
    const float* __restrict__ X0, const float* __restrict__ V0,
    const float* __restrict__ Y,  const float* __restrict__ theta,
    const float* __restrict__ W1, const float* __restrict__ b1,
    const float* __restrict__ W2, const float* __restrict__ b2,
    const float* __restrict__ noise, const int* __restrict__ obs_p,
    float* __restrict__ out)
{
    // Column-major (n-major, k-contiguous) weight panels -> direct B-fragment loads.
    __shared__ __align__(64) f16 sW1[H_DIM * KPAD];        // 128 KB: W1t[n][k], k<97 valid, rest 0
    __shared__ __align__(64) f16 sW2[D_DIM * H_DIM];       //  64 KB: W2t[d][h]
    __shared__ __align__(64) f16 sA [4][16 * KPAD];        //  16 KB: per-wave [s|X|Yt|0] panel
    __shared__ __align__(64) f16 sH [4][16 * HSTR];        //   5 KB: per-wave tanh staging (K=32)
    __shared__ __align__(64) float sX[4][16 * D_DIM];      //  16 KB: per-wave X state (fp32)
    __shared__ __align__(64) float sN[4][2][16 * D_DIM];   //  32 KB: double-buffered noise tiles
    __shared__ __align__(64) float sV[4][16];
    __shared__ __align__(64) float sb1[H_DIM];
    __shared__ __align__(64) float sb2[D_DIM];
    __shared__ __align__(64) float sth[D_DIM];

    const int tid  = threadIdx.x;
    const int w    = tid >> 5;
    const int lane = tid & 31;
    const int r0g  = blockIdx.x * 64 + w * 16;   // first global row of this wave's tile

    // ---- cooperative init: weights / biases into LDS ----
    for (int idx = tid; idx < H_DIM * KPAD; idx += 128) {
        int n = idx >> 7, k = idx & (KPAD - 1);
        sW1[idx] = (k < 97) ? (f16)W1[k * H_DIM + n] : (f16)0.0f;
    }
    for (int idx = tid; idx < D_DIM * H_DIM; idx += 128) {
        int d = idx >> 9, h = idx & (H_DIM - 1);
        sW2[idx] = (f16)W2[h * D_DIM + d];
    }
    for (int j = tid; j < H_DIM; j += 128) sb1[j] = b1[j];
    if (tid < D_DIM) { sb2[tid] = b2[tid]; sth[tid] = theta[tid]; }

    const int obs = obs_p[0];

    // ---- per-wave state init ----
    for (int idx = lane; idx < 16 * KPAD; idx += 32) sA[w][idx] = (f16)0.0f;
    for (int idx = lane; idx < 16 * D_DIM; idx += 32) {
        int r = idx >> 6, c = idx & 63;
        float x = X0[(size_t)(r0g + r) * D_DIM + c];
        sX[w][idx] = x;
        sA[w][r * KPAD + 1 + c] = (f16)x;
    }
    for (int idx = lane; idx < 16 * P_DIM; idx += 32) {
        int r = idx >> 5, p = idx & 31;   // Y shape (N, T=1, P)
        sA[w][r * KPAD + 65 + p] = (f16)Y[(size_t)(r0g + r) * P_DIM + obs * P_DIM + p];
    }
    if (lane < 16) sV[w][lane] = V0[r0g + lane];

    // prime the noise pipeline: step-0 tile into buffer 0
    const unsigned nbuf[2] = { (unsigned)(uintptr_t)&sN[w][0][0],
                               (unsigned)(uintptr_t)&sN[w][1][0] };
    const float* noise_tile0 = noise + (size_t)r0g * D_DIM;
    noise_prefetch(noise_tile0, nbuf[0], lane);

    __syncthreads();

    const int c16   = lane & 15;
    const int half8 = (lane < 16) ? 0 : 8;   // C-layout row half
    const int bsel  = (lane < 16) ? 0 : 16;  // B-frag K select
    const f16* aRow = &sA[w][c16 * KPAD];
    const f16* hRow = &sH[w][c16 * HSTR];

    #pragma unroll 1
    for (int m = 0; m < M_STEPS; ++m) {
        // time column (same value for every row of the tile)
        float s = DT_F * (float)m;
        if (lane < 16) sA[w][lane * KPAD] = (f16)s;

        // kick off next step's noise tile while the GEMMs run
        if (m + 1 < M_STEPS)
            noise_prefetch(noise + (size_t)(m + 1) * N_PART * D_DIM + (size_t)r0g * D_DIM,
                           nbuf[(m + 1) & 1], lane);

        v8f zacc[4] = {};   // Z tile accumulators (16 x 64)

        #pragma unroll 1
        for (int jp = 0; jp < 16; ++jp) {       // two H-tiles (32 hidden cols) per iter
            v8f acc0 = {}, acc1 = {};
            #pragma unroll
            for (int kc = 0; kc < 4; ++kc) {    // K = 128 (padded 97)
                v16h va  = a_frag(aRow, kc * 32 + bsel);
                v16h vb0 = *(const v16h*)(&sW1[(jp * 32      + c16) * KPAD + kc * 32 + bsel]);
                v16h vb1 = *(const v16h*)(&sW1[(jp * 32 + 16 + c16) * KPAD + kc * 32 + bsel]);
                acc0 = __builtin_amdgcn_wmma_f32_16x16x32_f16(false, va, false, vb0, (short)0, acc0, false, false);
                acc1 = __builtin_amdgcn_wmma_f32_16x16x32_f16(false, va, false, vb1, (short)0, acc1, false, false);
            }
            float bb0 = sb1[jp * 32 + c16];
            float bb1 = sb1[jp * 32 + 16 + c16];
            #pragma unroll
            for (int v = 0; v < 8; ++v) {       // bias + tanh, C-layout -> row-major staging
                sH[w][(v + half8) * HSTR +      c16] = (f16)fast_tanh(acc0[v] + bb0);
                sH[w][(v + half8) * HSTR + 16 + c16] = (f16)fast_tanh(acc1[v] + bb1);
            }
            v16h vh = a_frag(hRow, bsel);       // reload 16x32 h chunk as A fragment
            #pragma unroll
            for (int ot = 0; ot < 4; ++ot) {    // accumulate into Z (K chunk = jp)
                v16h vb = *(const v16h*)(&sW2[(ot * 16 + c16) * H_DIM + jp * 32 + bsel]);
                zacc[ot] = __builtin_amdgcn_wmma_f32_16x16x32_f16(false, vh, false, vb, (short)0, zacc[ot], false, false);
            }
        }

        // current noise tile must have landed (in-order async completion:
        // <=8 outstanding are the ones we just issued for step m+1)
        if (m + 1 < M_STEPS) asm volatile("s_wait_asynccnt 0x8" ::: "memory");
        else                 asm volatile("s_wait_asynccnt 0x0" ::: "memory");

        // ---- elementwise SDE update straight from C layout ----
        const float* nt = &sN[w][m & 1][0];
        float vp[8] = {0,0,0,0,0,0,0,0};
        #pragma unroll
        for (int ot = 0; ot < 4; ++ot) {
            int c = ot * 16 + c16;
            float th = sth[c], bb = sb2[c];
            #pragma unroll
            for (int v = 0; v < 8; ++v) {
                int r = v + half8;
                float z  = zacc[ot][v] + bb;
                float wm = SQRT_DT_F * nt[r * D_DIM + c];
                float x  = sX[w][r * D_DIM + c];
                float xn = x + DT_F * (th - x - z) + wm;      // SIGMA = 1
                sX[w][r * D_DIM + c] = xn;
                sA[w][r * KPAD + 1 + c] = (f16)xn;
                vp[v] += z * (wm - 0.5f * DT_F * z);          // dt*(-0.5 z^2) + z*wm
            }
        }
        #pragma unroll
        for (int v = 0; v < 8; ++v) atomicAdd(&sV[w][v + half8], vp[v]);  // ds_add_f32
    }

    // ---- epilogue: out = [X (N*64) | V (N)] ----
    for (int idx = lane; idx < 16 * D_DIM; idx += 32) {
        int r = idx >> 6;
        out[(size_t)(r0g + r) * D_DIM + (idx & 63)] = sX[w][idx];
    }
    if (lane < 16) out[(size_t)N_PART * D_DIM + r0g + lane] = sV[w][lane];
}

extern "C" void kernel_launch(void* const* d_in, const int* in_sizes, int n_in,
                              void* d_out, int out_size, void* d_ws, size_t ws_size,
                              hipStream_t stream) {
    const float* X0    = (const float*)d_in[0];
    const float* V0    = (const float*)d_in[1];
    const float* Y     = (const float*)d_in[2];
    const float* theta = (const float*)d_in[3];
    const float* W1    = (const float*)d_in[4];
    const float* b1    = (const float*)d_in[5];
    const float* W2    = (const float*)d_in[6];
    const float* b2    = (const float*)d_in[7];
    const float* noise = (const float*)d_in[8];
    const int*   obs   = (const int*)d_in[9];
    float* out = (float*)d_out;

    dim3 grid(N_PART / 64), block(128);
    hipLaunchKernelGGL(sde_wmma_kernel, grid, block, 0, stream,
                       X0, V0, Y, theta, W1, b1, W2, b2, noise, obs, out);
}
